// MoE_85822036508886
// MI455X (gfx1250) — compile-verified, matrix operand order checked by reading, not run on hardware
//
#include <hip/hip_runtime.h>
#include <cmath>
#include <cstdint>

// ---------------------------------------------------------------------------
// MoE (top-2 of 8 experts, D=512, F=1024) + shared GeluMLP for MI455X/gfx1250.
// bf16 WMMA (v_wmma_f32_16x16x32_bf16), wave32, fused fc1->gelu->fc2 with the
// hidden tile held in LDS. Sparse gather routing (4x less matrix math than the
// dense reference einsum; identical result since combine weights are 0 for
// unselected experts). x-tile gather uses GLOBAL_LOAD_ASYNC_TO_LDS_B128
// (ASYNCcnt) to avoid VGPR staging.
// ---------------------------------------------------------------------------

#define N_TOK 16384   // B*T*H*W
#define DDIM  512
#define FDIM  1024
#define NEXP  8
#define TM    64              // tokens per block tile (halves L2 weight traffic vs 32)
#define XP    (DDIM + 8)      // LDS pitch for x tile (bank-conflict padding)
#define HP    (FDIM + 8)      // LDS pitch for hidden tile

typedef __attribute__((ext_vector_type(16))) __bf16 v16bf;
typedef __attribute__((ext_vector_type(8)))  __bf16 bf16x8;
typedef __attribute__((ext_vector_type(4)))  __bf16 bf16x4;
typedef __attribute__((ext_vector_type(8)))  float  v8f;

static __device__ __forceinline__ v8f wmma_bf16(v16bf a, v16bf b, v8f c) {
    // (neg_a, A, neg_b, B, c_mod, C, reuse_a, reuse_b)
    return __builtin_amdgcn_wmma_f32_16x16x32_bf16(false, a, false, b,
                                                   (short)0, c, false, false);
}

// Load one 16x32 bf16 fragment (A layout; B uses the mirrored layout which is
// identical when the source is stored [N][K] row-major, i.e. B^T).
// lanes 0-15 : row = lane,     K = k0+0..7  and k0+16..23
// lanes 16-31: row = lane-16,  K = k0+8..15 and k0+24..31
static __device__ __forceinline__ v16bf frag_ld(const __bf16* tile, int pitch, int lane) {
    const int half = (lane >> 4) & 1;
    const int row  = lane & 15;
    const __bf16* p = tile + row * pitch + half * 8;
    bf16x8 lo = *reinterpret_cast<const bf16x8*>(p);
    bf16x8 hi = *reinterpret_cast<const bf16x8*>(p + 16);
    return __builtin_shufflevector(lo, hi, 0,1,2,3,4,5,6,7,
                                           8,9,10,11,12,13,14,15);
}

static __device__ __forceinline__ float gelu_exact(float x) {
    return 0.5f * x * (1.0f + erff(x * 0.70710678118654752f));
}

// --------------------------- small utility kernels -------------------------

__global__ void k_init(int* __restrict__ counts) {
    if (threadIdx.x < NEXP) counts[threadIdx.x] = 0;
}

__global__ void k_cvt_x(const float* __restrict__ in, __bf16* __restrict__ out) {
    const size_t i = ((size_t)blockIdx.x * 256 + threadIdx.x) * 4;
    const float4 v = *reinterpret_cast<const float4*>(in + i);
    bf16x4 o;
    o[0] = (__bf16)v.x; o[1] = (__bf16)v.y; o[2] = (__bf16)v.z; o[3] = (__bf16)v.w;
    *reinterpret_cast<bf16x4*>(out + i) = o;
}

// in: [G][R][C] f32  ->  out: [G][C][R] bf16   (LDS-tiled transpose+convert)
__global__ void k_tcvt(const float* __restrict__ in, __bf16* __restrict__ out,
                       int R, int C) {
    __shared__ float t[32][33];
    const int g  = blockIdx.z;
    const int rb = blockIdx.y * 32;
    const int cb = blockIdx.x * 32;
    const int tx = threadIdx.x & 31;
    const int ty = threadIdx.x >> 5;        // 0..7
    #pragma unroll
    for (int s = 0; s < 4; ++s) {
        const int r = ty + s * 8;
        t[r][tx] = in[((size_t)g * R + rb + r) * C + cb + tx];
    }
    __syncthreads();
    #pragma unroll
    for (int s = 0; s < 4; ++s) {
        const int c = ty + s * 8;
        out[((size_t)g * C + cb + c) * R + rb + tx] = (__bf16)t[tx][c];
    }
}

// ------------------------------- gate/router -------------------------------
// One wave per token. softmax->top2->renorm collapses to w1 = 1/(1+e^(v2-v1)).
__global__ void k_gate(const float* __restrict__ x, const float* __restrict__ gw,
                       int* __restrict__ counts, int* __restrict__ dlist,
                       float* __restrict__ wlist) {
    const int lane  = threadIdx.x & 31;
    const int wv    = threadIdx.x >> 5;
    const int token = blockIdx.x * 8 + wv;
    const float* xr = x + (size_t)token * DDIM;

    float acc[NEXP];
    #pragma unroll
    for (int e = 0; e < NEXP; ++e) acc[e] = 0.0f;
    for (int i = 0; i < DDIM / 32; ++i) {
        const int d = lane + i * 32;
        const float xv = xr[d];
        #pragma unroll
        for (int e = 0; e < NEXP; ++e) acc[e] += xv * gw[e * DDIM + d];
    }
    #pragma unroll
    for (int e = 0; e < NEXP; ++e) {
        #pragma unroll
        for (int off = 16; off > 0; off >>= 1)
            acc[e] += __shfl_xor(acc[e], off, 32);
    }
    if (lane == 0) {
        int i1 = 0; float v1 = acc[0];
        #pragma unroll
        for (int e = 1; e < NEXP; ++e) if (acc[e] > v1) { v1 = acc[e]; i1 = e; }
        int i2 = -1; float v2 = -3.4e38f;
        #pragma unroll
        for (int e = 0; e < NEXP; ++e)
            if (e != i1 && acc[e] > v2) { v2 = acc[e]; i2 = e; }
        const float w1 = 1.0f / (1.0f + expf(v2 - v1));
        const float w2 = 1.0f - w1;
        const int s1 = atomicAdd(&counts[i1], 1);
        dlist[i1 * N_TOK + s1] = token * 2 + 0;  wlist[i1 * N_TOK + s1] = w1;
        const int s2 = atomicAdd(&counts[i2], 1);
        dlist[i2 * N_TOK + s2] = token * 2 + 1;  wlist[i2 * N_TOK + s2] = w2;
    }
}

// --------------------------- fused expert MLP ------------------------------
// 256 threads = 8 waves. Tile: 64 gathered tokens -> fc1(512->1024)+gelu in
// LDS (bf16) -> fc2(1024->512) -> scaled scatter to ybuf[rowid*D].
// GATHER=true : rowid = token*2+k (per-assignment slot), weight from router.
// GATHER=false: rowid = token (shared expert, weight 1, writes d_out).
template<bool GATHER>
__global__ void k_mlp(const __bf16* __restrict__ xbf,
                      const __bf16* __restrict__ W1T, const float* __restrict__ b1,
                      const __bf16* __restrict__ W2T, const float* __restrict__ b2,
                      const int* __restrict__ counts,
                      const int* __restrict__ dlist,
                      const float* __restrict__ wlist,
                      float* __restrict__ ybuf) {
    const int e   = GATHER ? blockIdx.y : 0;
    const int cnt = GATHER ? counts[e] : N_TOK;
    const int m0  = blockIdx.x * TM;
    if (m0 >= cnt) return;                      // uniform: EXEC stays all-ones

    extern __shared__ char smem[];
    __bf16* xs = reinterpret_cast<__bf16*>(smem);          // [TM][XP]
    __bf16* hs = xs + TM * XP;                             // [TM][HP]
    int*    dl = reinterpret_cast<int*>(hs + TM * HP);     // [TM] dest rows
    float*  wl = reinterpret_cast<float*>(dl + TM);        // [TM] weights

    const int tid  = threadIdx.x;
    const int lane = tid & 31;
    const int wv   = tid >> 5;
    const int half = (lane >> 4) & 1;
    const int col  = lane & 15;

    if (tid < TM) {
        if (GATHER) {
            const int slot = m0 + tid;
            if (slot < cnt) { dl[tid] = dlist[e * N_TOK + slot];
                              wl[tid] = wlist[e * N_TOK + slot]; }
            else            { dl[tid] = -1; wl[tid] = 0.0f; }
        } else {
            dl[tid] = m0 + tid; wl[tid] = 1.0f;
        }
    }
    __syncthreads();

    // Gather 64 token rows (bf16) straight into LDS with async copies
    // (GLOBAL_LOAD_ASYNC_TO_LDS_B128, tracked by ASYNCcnt; LDS address is the
    // low 32 bits of the generic pointer per the ISA aperture rule).
    {
        const int row  = tid >> 2;              // 4 threads per row
        const int part = tid & 3;               // each covers 128 elements
        const int d0   = dl[row];
        __bf16* dst = xs + row * XP + part * 128;
        if (d0 >= 0) {
            const int tok = GATHER ? (d0 >> 1) : d0;
            const __bf16* src = xbf + (size_t)tok * DDIM + part * 128;
            #pragma unroll
            for (int j = 0; j < 16; ++j) {
                const unsigned ldsa = (unsigned)(uintptr_t)(dst + j * 8);
                asm volatile("global_load_async_to_lds_b128 %0, %1, off"
                             :: "v"(ldsa), "v"(src + j * 8) : "memory");
            }
        } else {
            bf16x8 z = {};
            #pragma unroll
            for (int j = 0; j < 16; ++j) reinterpret_cast<bf16x8*>(dst)[j] = z;
        }
    }
    asm volatile("s_wait_asynccnt 0x0" ::: "memory");
    __syncthreads();

    // ---------------- Phase 1: hid = gelu(x @ W1 + b1) ----------------
    const __bf16* W1e = W1T + (size_t)e * FDIM * DDIM;     // [F][D] (pre-transposed)
    const float*  b1e = b1 + (size_t)e * FDIM;
    for (int ft = 0; ft < 8; ++ft) {                       // 8 F-tiles per wave
        const int f0 = wv * (FDIM / 8) + ft * 16;
        const __bf16* bb = W1e + (size_t)f0 * DDIM;
        __builtin_prefetch(bb + (size_t)(16 + col) * DDIM + half * 8, 0, 1);
        v8f acc0 = {}, acc1 = {}, acc2 = {}, acc3 = {};
        for (int k0 = 0; k0 < DDIM; k0 += 32) {            // 16 WMMA steps x 4 M
            const v16bf bf = frag_ld(bb + k0, DDIM, lane);
            acc0 = wmma_bf16(frag_ld(xs +           k0, XP, lane), bf, acc0);
            acc1 = wmma_bf16(frag_ld(xs + 16 * XP + k0, XP, lane), bf, acc1);
            acc2 = wmma_bf16(frag_ld(xs + 32 * XP + k0, XP, lane), bf, acc2);
            acc3 = wmma_bf16(frag_ld(xs + 48 * XP + k0, XP, lane), bf, acc3);
        }
        const float bias = b1e[f0 + col];
        #pragma unroll
        for (int r = 0; r < 8; ++r) {                      // D layout: M=r+half*8
            const int m = r + half * 8;
            hs[(size_t)m        * HP + f0 + col] = (__bf16)gelu_exact(acc0[r] + bias);
            hs[(size_t)(m + 16) * HP + f0 + col] = (__bf16)gelu_exact(acc1[r] + bias);
            hs[(size_t)(m + 32) * HP + f0 + col] = (__bf16)gelu_exact(acc2[r] + bias);
            hs[(size_t)(m + 48) * HP + f0 + col] = (__bf16)gelu_exact(acc3[r] + bias);
        }
    }
    __syncthreads();

    // ---------------- Phase 2: y = (hid @ W2 + b2) * w, scatter -------
    const __bf16* W2e = W2T + (size_t)e * DDIM * FDIM;     // [D][F] (pre-transposed)
    const float*  b2e = b2 + (size_t)e * DDIM;
    for (int nt = 0; nt < 4; ++nt) {                       // 4 D-tiles per wave
        const int n0 = wv * (DDIM / 8) + nt * 16;
        const __bf16* bb = W2e + (size_t)n0 * FDIM;
        __builtin_prefetch(bb + (size_t)(16 + col) * FDIM + half * 8, 0, 1);
        v8f acc0 = {}, acc1 = {}, acc2 = {}, acc3 = {};
        for (int k0 = 0; k0 < FDIM; k0 += 32) {            // 32 WMMA steps x 4 M
            const v16bf bf = frag_ld(bb + k0, FDIM, lane);
            acc0 = wmma_bf16(frag_ld(hs +           k0, HP, lane), bf, acc0);
            acc1 = wmma_bf16(frag_ld(hs + 16 * HP + k0, HP, lane), bf, acc1);
            acc2 = wmma_bf16(frag_ld(hs + 32 * HP + k0, HP, lane), bf, acc2);
            acc3 = wmma_bf16(frag_ld(hs + 48 * HP + k0, HP, lane), bf, acc3);
        }
        const float bias = b2e[n0 + col];
        #pragma unroll
        for (int r = 0; r < 8; ++r) {
            const int m = r + half * 8;
            const int d0 = dl[m],      d1 = dl[m + 16];
            const int d2 = dl[m + 32], d3 = dl[m + 48];
            if (d0 >= 0)
                ybuf[(size_t)d0 * DDIM + n0 + col] = (acc0[r] + bias) * wl[m];
            if (d1 >= 0)
                ybuf[(size_t)d1 * DDIM + n0 + col] = (acc1[r] + bias) * wl[m + 16];
            if (d2 >= 0)
                ybuf[(size_t)d2 * DDIM + n0 + col] = (acc2[r] + bias) * wl[m + 32];
            if (d3 >= 0)
                ybuf[(size_t)d3 * DDIM + n0 + col] = (acc3[r] + bias) * wl[m + 48];
        }
    }
}

// out[n] += y_ws[2n] + y_ws[2n+1]   (shared result already in out)
__global__ void k_combine(const float* __restrict__ yws, float* __restrict__ out) {
    const int i = blockIdx.x * 256 + threadIdx.x;          // over N*D/4
    const int q = i & (DDIM / 4 - 1);
    const int n = i >> 7;                                  // D/4 == 128
    const float4* y = reinterpret_cast<const float4*>(yws);
    float4* o = reinterpret_cast<float4*>(out);
    const float4 a = y[(size_t)(2 * n)     * (DDIM / 4) + q];
    const float4 b = y[(size_t)(2 * n + 1) * (DDIM / 4) + q];
    const float4 c = o[i];
    o[i] = make_float4(c.x + a.x + b.x, c.y + a.y + b.y,
                       c.z + a.z + b.z, c.w + a.w + b.w);
}

// ------------------------------- launcher ----------------------------------

extern "C" void kernel_launch(void* const* d_in, const int* in_sizes, int n_in,
                              void* d_out, int out_size, void* d_ws, size_t ws_size,
                              hipStream_t stream) {
    (void)in_sizes; (void)n_in; (void)out_size; (void)ws_size;
    const float* x   = (const float*)d_in[0];
    const float* gw  = (const float*)d_in[1];
    const float* W1  = (const float*)d_in[2];
    const float* b1  = (const float*)d_in[3];
    const float* W2  = (const float*)d_in[4];
    const float* b2  = (const float*)d_in[5];
    const float* Ws1 = (const float*)d_in[6];
    const float* bs1 = (const float*)d_in[7];
    const float* Ws2 = (const float*)d_in[8];
    const float* bs2 = (const float*)d_in[9];
    float* out = (float*)d_out;

    char* p = (char*)d_ws;
    auto carve = [&](size_t bytes) {
        char* q = p; p += (bytes + 255) & ~(size_t)255; return q;
    };
    __bf16* xbf    = (__bf16*)carve((size_t)N_TOK * DDIM * 2);          // 16 MB
    __bf16* W1T    = (__bf16*)carve((size_t)NEXP * FDIM * DDIM * 2);    //  8 MB
    __bf16* W2T    = (__bf16*)carve((size_t)NEXP * DDIM * FDIM * 2);    //  8 MB
    __bf16* Ws1T   = (__bf16*)carve((size_t)FDIM * DDIM * 2);           //  1 MB
    __bf16* Ws2T   = (__bf16*)carve((size_t)DDIM * FDIM * 2);           //  1 MB
    int*    counts = (int*)   carve(NEXP * sizeof(int));
    int*    dlist  = (int*)   carve((size_t)NEXP * N_TOK * sizeof(int));
    float*  wlist  = (float*) carve((size_t)NEXP * N_TOK * sizeof(float));
    float*  yws    = (float*) carve((size_t)N_TOK * 2 * DDIM * sizeof(float)); // 64 MB

    const size_t smem = (size_t)TM * XP * 2 + (size_t)TM * HP * 2
                      + TM * sizeof(int) + TM * sizeof(float);          // ~195 KB

    hipLaunchKernelGGL(k_init, dim3(1), dim3(32), 0, stream, counts);
    hipLaunchKernelGGL(k_cvt_x, dim3((N_TOK * (size_t)DDIM) / 1024), dim3(256),
                       0, stream, x, xbf);
    hipLaunchKernelGGL(k_tcvt, dim3(FDIM / 32, DDIM / 32, NEXP), dim3(256),
                       0, stream, W1, W1T, DDIM, FDIM);                 // [E][D][F]->[E][F][D]
    hipLaunchKernelGGL(k_tcvt, dim3(DDIM / 32, FDIM / 32, NEXP), dim3(256),
                       0, stream, W2, W2T, FDIM, DDIM);                 // [E][F][D]->[E][D][F]
    hipLaunchKernelGGL(k_tcvt, dim3(FDIM / 32, DDIM / 32, 1), dim3(256),
                       0, stream, Ws1, Ws1T, DDIM, FDIM);
    hipLaunchKernelGGL(k_tcvt, dim3(DDIM / 32, FDIM / 32, 1), dim3(256),
                       0, stream, Ws2, Ws2T, FDIM, DDIM);
    hipLaunchKernelGGL(k_gate, dim3(N_TOK / 8), dim3(256), 0, stream,
                       x, gw, counts, dlist, wlist);
    hipLaunchKernelGGL((k_mlp<true>), dim3(N_TOK / TM, NEXP), dim3(256), smem,
                       stream, xbf, W1T, b1, W2T, b2, counts, dlist, wlist, yws);
    hipLaunchKernelGGL((k_mlp<false>), dim3(N_TOK / TM, 1), dim3(256), smem,
                       stream, xbf, Ws1T, bs1, Ws2T, bs2,
                       (const int*)nullptr, (const int*)nullptr,
                       (const float*)nullptr, out);
    hipLaunchKernelGGL(k_combine, dim3((N_TOK * (size_t)DDIM) / 1024), dim3(256),
                       0, stream, yws, out);
}
